// MultiScaleRetention_91096256348590
// MI455X (gfx1250) — matrix operand order repaired and checked by reading, not compiled, not run on hardware
//
#include <hip/hip_runtime.h>

#define Bx 2
#define Tt 2048
#define Cc 1024
#define NH 16
#define Dd 64

typedef __attribute__((ext_vector_type(16))) __bf16 v16bf;
typedef __attribute__((ext_vector_type(8)))  float  v8f;

// ---------- async global->LDS (CDNA5 GLOBAL_LOAD_ASYNC_TO_LDS_B128, ASYNCcnt) ----------
#if defined(__has_builtin)
#if __has_builtin(__builtin_amdgcn_global_load_async_to_lds_b128) && \
    __has_builtin(__builtin_amdgcn_s_wait_asynccnt)
#define HAVE_ASYNC_LDS 1
#endif
#endif

#ifdef HAVE_ASYNC_LDS
typedef int v4i_ __attribute__((vector_size(16)));
typedef __attribute__((address_space(1))) v4i_* gp128;   // global 16B-vector ptr
typedef __attribute__((address_space(3))) v4i_* lp128;   // LDS 16B-vector ptr
__device__ __forceinline__ void async_copy_b128(const void* gsrc, void* ldst) {
    __builtin_amdgcn_global_load_async_to_lds_b128(
        (gp128)(uintptr_t)gsrc,
        (lp128)(uint32_t)(uintptr_t)ldst, 0, 0);
}
__device__ __forceinline__ void async_wait0() { __builtin_amdgcn_s_wait_asynccnt(0); }
#endif

// ---------- fragment loaders (wave32 WMMA, contiguous k = 16*(lane/16) + e in both A and B) ----------

// A matrix 16x32 bf16, row-major LDS tile (row m, col k), row stride lda elems.
__device__ __forceinline__ v16bf load_a_frag(const __bf16* lds, int lda, int lane, int kk) {
    int m  = lane & 15;
    int h16 = (lane >> 4) << 4;        // 0 or 16
    const __bf16* r = lds + m * lda;
    v16bf a;
#pragma unroll
    for (int e = 0; e < 16; ++e) a[e] = r[kk + h16 + e];
    return a;
}

// B matrix 32x16 bf16 where LDS tile is stored (N rows x K cols) row-major
// (i.e. B[k][n] = tile[n][k]); lane n = lane%16, K = kk + 16*(lane/16) + e.
__device__ __forceinline__ v16bf load_b_frag_nk(const __bf16* lds, int ldn, int lane,
                                                int ncol_base, int kk) {
    int n  = ncol_base + (lane & 15);
    int kb = kk + ((lane >> 4) << 4);
    const __bf16* r = lds + n * ldn + kb;
    v16bf b;
#pragma unroll
    for (int e = 0; e < 16; ++e) b[e] = r[e];
    return b;
}

// B matrix 32x16 bf16 where LDS tile is stored (K rows x N cols) row-major
// (B[k][n] = tile[k][n]); gather down column n.
__device__ __forceinline__ v16bf load_b_frag_kn(const __bf16* lds, int ldn, int lane,
                                                int ncol_base, int kk) {
    int n  = ncol_base + (lane & 15);
    int kb = kk + ((lane >> 4) << 4);
    v16bf b;
#pragma unroll
    for (int e = 0; e < 16; ++e) b[e] = lds[(size_t)(kb + e) * ldn + n];
    return b;
}

__device__ __forceinline__ v8f wmma_bf16(v16bf a, v16bf b, v8f c) {
    return __builtin_amdgcn_wmma_f32_16x16x32_bf16(false, a, false, b, (short)0, c, false, false);
}

// ---------- fp32 -> bf16 ----------
__global__ __launch_bounds__(256) void k_cvt_bf16(const float* __restrict__ s,
                                                  __bf16* __restrict__ d, size_t n) {
    size_t i = (size_t)blockIdx.x * blockDim.x + threadIdx.x;
    size_t stride = (size_t)gridDim.x * blockDim.x;
    for (; i < n; i += stride) d[i] = (__bf16)s[i];
}

__global__ void k_zero_stats(float* stats) {
    if (threadIdx.x < 64) stats[threadIdx.x] = 0.0f;
}

// ---------- bf16 GEMM:  C(M,N) = A(M,K) * W(N,K)^T , fp32 out ----------
// 256 threads, 128x128 block tile, 8 waves in 4x2 (waveM x waveN) layout,
// each wave computes 32x64 (2 A-frags x 4 B-frags = 8 WMMA / K-step),
// double-buffered LDS, async global->LDS staging, one barrier per K-step.
__global__ __launch_bounds__(256) void k_gemm_bf16(const __bf16* __restrict__ A,
                                                   const __bf16* __restrict__ W,
                                                   float* __restrict__ Co,
                                                   int M, int N, int K) {
    __shared__ __bf16 As[2][128][40];
    __shared__ __bf16 Ws[2][128][40];
    int tid  = threadIdx.x;
    int lane = tid & 31;
    int wave = tid >> 5;
    int wm = (wave & 3) << 5;     // 0,32,64,96
    int wn = (wave >> 2) << 6;    // 0,64
    int m0 = blockIdx.x * 128;
    int n0 = blockIdx.y * 128;

    int lrow = tid >> 1;                // 0..127
    int lcol = (tid & 1) << 4;          // 0 or 16
    const __bf16* aptr = A + (size_t)(m0 + lrow) * K + lcol;
    const __bf16* wptr = W + (size_t)(n0 + lrow) * K + lcol;

    v8f acc[2][4];
#pragma unroll
    for (int a = 0; a < 2; ++a)
#pragma unroll
        for (int f = 0; f < 4; ++f) acc[a][f] = (v8f){0,0,0,0,0,0,0,0};

    // prologue: stage K-tile 0 into buffer 0
#ifdef HAVE_ASYNC_LDS
    async_copy_b128(aptr,     &As[0][lrow][lcol]);
    async_copy_b128(aptr + 8, &As[0][lrow][lcol + 8]);
    async_copy_b128(wptr,     &Ws[0][lrow][lcol]);
    async_copy_b128(wptr + 8, &Ws[0][lrow][lcol + 8]);
    async_wait0();
#else
    {
        const uint4* as = reinterpret_cast<const uint4*>(aptr);
        const uint4* wsrc = reinterpret_cast<const uint4*>(wptr);
        uint4 a0 = as[0], a1 = as[1], w0 = wsrc[0], w1 = wsrc[1];
        *reinterpret_cast<uint4*>(&As[0][lrow][lcol])     = a0;
        *reinterpret_cast<uint4*>(&As[0][lrow][lcol + 8]) = a1;
        *reinterpret_cast<uint4*>(&Ws[0][lrow][lcol])     = w0;
        *reinterpret_cast<uint4*>(&Ws[0][lrow][lcol + 8]) = w1;
    }
#endif
    __syncthreads();

    for (int k0 = 0; k0 < K; k0 += 32) {
        int  cur = (k0 >> 5) & 1;
        int  nb  = cur ^ 1;
        bool hn  = (k0 + 32) < K;

#ifdef HAVE_ASYNC_LDS
        // fire-and-forget staging of tile k+1 straight into LDS (ASYNCcnt)
        if (hn) {
            async_copy_b128(aptr + k0 + 32,     &As[nb][lrow][lcol]);
            async_copy_b128(aptr + k0 + 32 + 8, &As[nb][lrow][lcol + 8]);
            async_copy_b128(wptr + k0 + 32,     &Ws[nb][lrow][lcol]);
            async_copy_b128(wptr + k0 + 32 + 8, &Ws[nb][lrow][lcol + 8]);
            if (k0 + 64 < K) {   // L2 prefetch two tiles ahead
                __builtin_prefetch(aptr + k0 + 64, 0, 1);
                __builtin_prefetch(wptr + k0 + 64, 0, 1);
            }
        }
#else
        uint4 na0, na1, nw0, nw1;
        if (hn) {
            const uint4* as = reinterpret_cast<const uint4*>(aptr + k0 + 32);
            const uint4* wsrc = reinterpret_cast<const uint4*>(wptr + k0 + 32);
            na0 = as[0]; na1 = as[1]; nw0 = wsrc[0]; nw1 = wsrc[1];
            if (k0 + 64 < K) {
                __builtin_prefetch(aptr + k0 + 64, 0, 1);
                __builtin_prefetch(wptr + k0 + 64, 0, 1);
            }
        }
#endif

        // compute: 2 A-frags x 4 B-frags
        v16bf af0 = load_a_frag(&As[cur][wm][0],      40, lane, 0);
        v16bf af1 = load_a_frag(&As[cur][wm + 16][0], 40, lane, 0);
#pragma unroll
        for (int f = 0; f < 4; ++f) {
            v16bf bf_ = load_b_frag_nk(&Ws[cur][0][0], 40, lane, wn + f * 16, 0);
            acc[0][f] = wmma_bf16(af0, bf_, acc[0][f]);
            acc[1][f] = wmma_bf16(af1, bf_, acc[1][f]);
        }

#ifdef HAVE_ASYNC_LDS
        if (hn) async_wait0();
#else
        if (hn) {
            *reinterpret_cast<uint4*>(&As[nb][lrow][lcol])     = na0;
            *reinterpret_cast<uint4*>(&As[nb][lrow][lcol + 8]) = na1;
            *reinterpret_cast<uint4*>(&Ws[nb][lrow][lcol])     = nw0;
            *reinterpret_cast<uint4*>(&Ws[nb][lrow][lcol + 8]) = nw1;
        }
#endif
        __syncthreads();
    }

    // epilogue
    int nc = n0 + wn + (lane & 15);
#pragma unroll
    for (int a = 0; a < 2; ++a) {
        int mrow = m0 + wm + a * 16 + ((lane >> 4) << 3);
#pragma unroll
        for (int r = 0; r < 8; ++r) {
            size_t ro = (size_t)(mrow + r) * N + nc;
            Co[ro]      = acc[a][0][r];
            Co[ro + 16] = acc[a][1][r];
            Co[ro + 32] = acc[a][2][r];
            Co[ro + 48] = acc[a][3][r];
        }
    }
}

// ---------- RoPE/xPos + head split (fp32 qkv -> bf16 head-major q,k,v) ----------
__global__ __launch_bounds__(256) void k_rope(const float* __restrict__ qkv,
                                              __bf16* __restrict__ qb,
                                              __bf16* __restrict__ kb,
                                              __bf16* __restrict__ vb) {
    size_t gid = (size_t)blockIdx.x * blockDim.x + threadIdx.x;
    if (gid >= (size_t)Bx * Tt * NH * (Dd / 2)) return;
    int i = gid & 31;
    int h = (gid >> 5) & 15;
    int t = (gid >> 9) & (Tt - 1);
    int b = (int)(gid >> 20);

    const float* row = qkv + ((size_t)b * Tt + t) * (3 * Cc);
    int c = h * Dd + 2 * i;
    float q1 = row[c],            q2 = row[c + 1];
    float k1 = row[Cc + c],       k2 = row[Cc + c + 1];
    float v1 = row[2 * Cc + c],   v2 = row[2 * Cc + c + 1];

    float fi = 2.0f * (float)i;
    float inv_freq = exp2f(-(fi / (float)Dd) * 13.2877123795f); // log2(10000)
    float fr = (float)t * inv_freq;
    float cs = cosf(fr), sn = sinf(fr);
    float sbase = (fi + 0.4f * Dd) / (1.4f * Dd);
    float power = ((float)t - (float)(Tt / 2)) * (1.0f / 512.0f);
    float sc  = exp2f(power * log2f(sbase));
    float isc = 1.0f / sc;

    float qo1 = q1 * cs * sc - q2 * sn * sc;
    float qo2 = q2 * cs * sc + q1 * sn * sc;
    float ko1 = k1 * cs * isc - k2 * sn * isc;
    float ko2 = k2 * cs * isc + k1 * sn * isc;

    size_t o = ((size_t)(b * NH + h) * Tt + t) * Dd + 2 * i;
    qb[o] = (__bf16)qo1; qb[o + 1] = (__bf16)qo2;
    kb[o] = (__bf16)ko1; kb[o + 1] = (__bf16)ko2;
    vb[o] = (__bf16)v1;  vb[o + 1] = (__bf16)v2;
}

// ---------- retention: one wave per (b*h, 16-row t tile) ----------
__global__ __launch_bounds__(32) void k_retention(const __bf16* __restrict__ qb,
                                                  const __bf16* __restrict__ kb,
                                                  const __bf16* __restrict__ vb,
                                                  float* __restrict__ y,
                                                  float* __restrict__ stats) {
    __shared__ __bf16 Qs[16][72];
    __shared__ __bf16 Ks[32][72];
    __shared__ __bf16 Vs[32][72];
    __shared__ __bf16 Ss[16][40];

    int bh = blockIdx.x;            // 0..31
    int h  = bh & (NH - 1);
    int t0 = blockIdx.y * 16;
    int lane = threadIdx.x;

    const __bf16* qp = qb + (size_t)bh * Tt * Dd;
    const __bf16* kp = kb + (size_t)bh * Tt * Dd;
    const __bf16* vp = vb + (size_t)bh * Tt * Dd;

    // load the 16x64 q tile once
    {
        const __bf16* qsrc = qp + (size_t)(t0 + (lane >> 1)) * Dd + ((lane & 1) << 5);
        __bf16* qdst = &Qs[lane >> 1][(lane & 1) << 5];
#ifdef HAVE_ASYNC_LDS
#pragma unroll
        for (int j = 0; j < 4; ++j) async_copy_b128(qsrc + j * 8, qdst + j * 8);
        async_wait0();
#else
        const uint4* qs4 = reinterpret_cast<const uint4*>(qsrc);
#pragma unroll
        for (int j = 0; j < 4; ++j) *reinterpret_cast<uint4*>(qdst + j * 8) = qs4[j];
#endif
    }
    __syncthreads();
    v16bf aq0 = load_a_frag(&Qs[0][0], 72, lane, 0);
    v16bf aq1 = load_a_frag(&Qs[0][0], 72, lane, 32);

    float gamma = 1.0f - exp2f(-5.0f - (float)h);
    float lg = log2f(gamma);

    v8f y0 = {0,0,0,0,0,0,0,0}, y1 = y0, y2 = y0, y3 = y0;

    for (int s0 = 0; s0 <= t0 + 15; s0 += 32) {
        // stage 32x64 k and v chunks
        const __bf16* ksrc = kp + (size_t)(s0 + lane) * Dd;
        const __bf16* vsrc = vp + (size_t)(s0 + lane) * Dd;
#ifdef HAVE_ASYNC_LDS
#pragma unroll
        for (int j = 0; j < 8; ++j) {
            async_copy_b128(ksrc + j * 8, &Ks[lane][j * 8]);
            async_copy_b128(vsrc + j * 8, &Vs[lane][j * 8]);
        }
        async_wait0();
#else
        const uint4* k4 = reinterpret_cast<const uint4*>(ksrc);
        const uint4* v4 = reinterpret_cast<const uint4*>(vsrc);
#pragma unroll
        for (int j = 0; j < 8; ++j) {
            *reinterpret_cast<uint4*>(&Ks[lane][j * 8]) = k4[j];
            *reinterpret_cast<uint4*>(&Vs[lane][j * 8]) = v4[j];
        }
#endif
        __syncthreads();

        // scores = q . k^T over the 32 columns of this chunk
#pragma unroll
        for (int half = 0; half < 2; ++half) {
            v8f sacc = {0,0,0,0,0,0,0,0};
            sacc = wmma_bf16(aq0, load_b_frag_nk(&Ks[0][0], 72, lane, half * 16, 0),  sacc);
            sacc = wmma_bf16(aq1, load_b_frag_nk(&Ks[0][0], 72, lane, half * 16, 32), sacc);
            int tr = (lane >> 4) << 3;
            int scl = half * 16 + (lane & 15);
            int sg  = s0 + scl;
#pragma unroll
            for (int r = 0; r < 8; ++r) {
                int tg = t0 + tr + r;
                int df = tg - sg;
                float m = (df >= 0) ? exp2f((float)df * lg) * 0.125f : 0.0f;
                Ss[tr + r][scl] = (__bf16)(sacc[r] * m);
            }
        }
        __syncthreads();

        // y += masked_scores(16x32) . v(32x64)
        v16bf as = load_a_frag(&Ss[0][0], 40, lane, 0);
        y0 = wmma_bf16(as, load_b_frag_kn(&Vs[0][0], 72, lane,  0, 0), y0);
        y1 = wmma_bf16(as, load_b_frag_kn(&Vs[0][0], 72, lane, 16, 0), y1);
        y2 = wmma_bf16(as, load_b_frag_kn(&Vs[0][0], 72, lane, 32, 0), y2);
        y3 = wmma_bf16(as, load_b_frag_kn(&Vs[0][0], 72, lane, 48, 0), y3);
        __syncthreads();
    }

    // store y and accumulate groupnorm stats
    float lsum = 0.0f, lsq = 0.0f;
    int trb = t0 + ((lane >> 4) << 3);
    int dc  = lane & 15;
#pragma unroll
    for (int r = 0; r < 8; ++r) {
        size_t o = (size_t)bh * Tt * Dd + (size_t)(trb + r) * Dd + dc;
        float a = y0[r], b2 = y1[r], c2 = y2[r], d2 = y3[r];
        y[o]      = a;  y[o + 16] = b2;
        y[o + 32] = c2; y[o + 48] = d2;
        lsum += a + b2 + c2 + d2;
        lsq  += a * a + b2 * b2 + c2 * c2 + d2 * d2;
    }
#pragma unroll
    for (int off = 16; off > 0; off >>= 1) {
        lsum += __shfl_down(lsum, off, 32);
        lsq  += __shfl_down(lsq,  off, 32);
    }
    if (lane == 0) {
        atomicAdd(&stats[bh], lsum);
        atomicAdd(&stats[32 + bh], lsq);
    }
}

// ---------- groupnorm + silu gate -> bf16 ----------
__global__ __launch_bounds__(256) void k_gn_gate(const float* __restrict__ y,
                                                 const float* __restrict__ stats,
                                                 const float* __restrict__ g,
                                                 const float* __restrict__ gnw,
                                                 const float* __restrict__ gnb,
                                                 __bf16* __restrict__ gated) {
    size_t gid = (size_t)blockIdx.x * blockDim.x + threadIdx.x;
    if (gid >= (size_t)Bx * Tt * Cc) return;
    int c = gid & (Cc - 1);
    int t = (gid >> 10) & (Tt - 1);
    int b = (int)(gid >> 21);
    int hh = c >> 6, d = c & 63;
    int bh = b * NH + hh;

    float n    = (float)Dd * (float)Tt;
    float mean = stats[bh] / n;
    float var  = stats[32 + bh] / n - mean * mean;
    float yv   = y[((size_t)bh * Tt + t) * Dd + d];
    float yn   = (yv - mean) * rsqrtf(var + 1e-5f) * gnw[c] + gnb[c];
    float gv   = g[gid];
    float si   = gv / (1.0f + expf(-gv));
    gated[gid] = (__bf16)(si * yn);
}

// ---------------------------------------------------------------------------

extern "C" void kernel_launch(void* const* d_in, const int* in_sizes, int n_in,
                              void* d_out, int out_size, void* d_ws, size_t ws_size,
                              hipStream_t stream) {
    const float* x       = (const float*)d_in[0];
    const float* w_qkv   = (const float*)d_in[1];
    const float* w_gated = (const float*)d_in[2];
    const float* w_proj  = (const float*)d_in[3];
    const float* gn_w    = (const float*)d_in[4];
    const float* gn_b    = (const float*)d_in[5];
    float* out = (float*)d_out;

    const size_t M = (size_t)Bx * Tt;          // 4096
    char* ws = (char*)d_ws;
    size_t off = 0;
    auto carve = [&](size_t bytes) {
        void* p = ws + off;
        off = (off + bytes + 255) & ~(size_t)255;
        return p;
    };
    __bf16* x_bf    = (__bf16*)carve(M * Cc * 2);
    __bf16* wqkv_bf = (__bf16*)carve((size_t)3 * Cc * Cc * 2);
    __bf16* wg_bf   = (__bf16*)carve((size_t)Cc * Cc * 2);
    __bf16* wp_bf   = (__bf16*)carve((size_t)Cc * Cc * 2);
    float*  qkv_f   = (float*)carve(M * 3 * Cc * 4);
    __bf16* qb      = (__bf16*)carve(M * Cc * 2);
    __bf16* kb      = (__bf16*)carve(M * Cc * 2);
    __bf16* vb      = (__bf16*)carve(M * Cc * 2);
    float*  y_f     = (float*)carve(M * Cc * 4);
    float*  g_f     = (float*)carve(M * Cc * 4);
    __bf16* gated   = (__bf16*)carve(M * Cc * 2);
    float*  stats   = (float*)carve(64 * 4);

    // conversions
    k_cvt_bf16<<<2048, 256, 0, stream>>>(x, x_bf, M * Cc);
    k_cvt_bf16<<<2048, 256, 0, stream>>>(w_qkv, wqkv_bf, (size_t)3 * Cc * Cc);
    k_cvt_bf16<<<1024, 256, 0, stream>>>(w_gated, wg_bf, (size_t)Cc * Cc);
    k_cvt_bf16<<<1024, 256, 0, stream>>>(w_proj, wp_bf, (size_t)Cc * Cc);
    k_zero_stats<<<1, 64, 0, stream>>>(stats);

    // qkv = x @ w_qkv^T   (4096 x 3072)
    k_gemm_bf16<<<dim3(M / 128, (3 * Cc) / 128), 256, 0, stream>>>(
        x_bf, wqkv_bf, qkv_f, (int)M, 3 * Cc, Cc);
    // g = x @ w_gated^T   (4096 x 1024)
    k_gemm_bf16<<<dim3(M / 128, Cc / 128), 256, 0, stream>>>(
        x_bf, wg_bf, g_f, (int)M, Cc, Cc);

    // rope + head split
    k_rope<<<(Bx * Tt * NH * (Dd / 2)) / 256, 256, 0, stream>>>(qkv_f, qb, kb, vb);

    // retention
    k_retention<<<dim3(Bx * NH, Tt / 16), 32, 0, stream>>>(qb, kb, vb, y_f, stats);

    // groupnorm + silu gate
    k_gn_gate<<<(Bx * Tt * Cc) / 256, 256, 0, stream>>>(y_f, stats, g_f, gn_w, gn_b, gated);

    // out = gated @ w_proj^T
    k_gemm_bf16<<<dim3(M / 128, Cc / 128), 256, 0, stream>>>(
        gated, wp_bf, out, (int)M, Cc, Cc);
}
